// GNN_50629074485343
// MI455X (gfx1250) — matrix operand order repaired and checked by reading, not compiled, not run on hardware
//
#include <hip/hip_runtime.h>

typedef __attribute__((ext_vector_type(16))) _Float16 v16h;
typedef __attribute__((ext_vector_type(8)))  _Float16 v8h;
typedef __attribute__((ext_vector_type(8)))  float    v8f;

#define HFEAT 64
#define EPSV 1e-16f

// ---------------------------------------------------------------------------
// wd_ad[k] = dot(Wd[k,:], a_d)   (din x 64 -> din)  [folds hd GEMM into GEMV]
// ---------------------------------------------------------------------------
__global__ void gat_fuse_wdad(const float* __restrict__ Wd,
                              const float* __restrict__ a_d,
                              float* __restrict__ wdad, int din) {
  int k = blockIdx.x * blockDim.x + threadIdx.x;
  if (k < din) {
    float s = 0.f;
    #pragma unroll 8
    for (int j = 0; j < HFEAT; ++j) s += Wd[k * HFEAT + j] * a_d[j];
    wdad[k] = s;
  }
}

// ---------------------------------------------------------------------------
// hs = X @ Ws via V_WMMA_F32_16X16X32_F16, DIN compile-time -> K loop unrolls.
// Block: 256 threads = 8 waves. Wave w -> M-tile (blockIdx*2 + w/4), N-tile w%4.
// Ws staged TRANSPOSED in LDS (f16, padded stride) so each lane's B operand is
// two 16-byte runs 16 halves apart -> 2x ds_load_b128 per WMMA.
// Operand packing per CDNA5 ISA 7.12.2 (16-bit A 16x32 / B 32x16 / f32 C 16x16).
// ---------------------------------------------------------------------------
template <int DIN>
__global__ void gat_gemm_hs(const float* __restrict__ X,
                            const float* __restrict__ Ws,
                            float* __restrict__ hs, int n_rows) {
  constexpr int LSTR = DIN + 8;                 // pad: lane stride 4 banks, no conflicts
  __shared__ _Float16 sWt[HFEAT * LSTR];        // W^T as f16
  int tid = threadIdx.x;
  for (int i = tid; i < DIN * HFEAT; i += 256) {
    int k = i >> 6, c = i & 63;                 // Ws row-major [DIN][64]
    sWt[c * LSTR + k] = (_Float16)Ws[i];
  }
  __syncthreads();

  int wave = tid >> 5;
  int lane = tid & 31;
  int row0 = (blockIdx.x * 2 + (wave >> 2)) * 16;
  int nt   = wave & 3;
  int half = lane >> 4;                         // 0: lanes 0-15, 1: lanes 16-31
  int lrow = row0 + (lane & 15);
  if (lrow >= n_rows) lrow = n_rows - 1;        // clamp; EXEC must stay all-ones
  int col  = nt * 16 + (lane & 15);

  const float* xbase = X + (size_t)lrow * DIN;
  const _Float16* wbase = sWt + (size_t)col * LSTR;

  v8f acc = {};
  #pragma unroll
  for (int kb = 0; kb < DIN; kb += 32) {
    // A: lanes 0-15: K kb+0..7 (a[0..7]) and kb+16..23 (a[8..15]); lanes 16-31: +8
    const float4* xr = (const float4*)(xbase + kb + half * 8);   // 32B aligned
    float4 x0 = xr[0], x1 = xr[1], x2 = xr[4], x3 = xr[5];
    v16h a;
    a[0]=(_Float16)x0.x; a[1]=(_Float16)x0.y; a[2]=(_Float16)x0.z; a[3]=(_Float16)x0.w;
    a[4]=(_Float16)x1.x; a[5]=(_Float16)x1.y; a[6]=(_Float16)x1.z; a[7]=(_Float16)x1.w;
    a[8]=(_Float16)x2.x; a[9]=(_Float16)x2.y; a[10]=(_Float16)x2.z; a[11]=(_Float16)x2.w;
    a[12]=(_Float16)x3.x; a[13]=(_Float16)x3.y; a[14]=(_Float16)x3.z; a[15]=(_Float16)x3.w;
    // B: lanes 0-15: col=lane, K kb+0..7 and kb+16..23; lanes 16-31: col=lane-16, +8
    const _Float16* wc = wbase + kb + half * 8;                  // 16B aligned
    v8h b0 = *(const v8h*)wc;
    v8h b1 = *(const v8h*)(wc + 16);
    v16h b;
    #pragma unroll
    for (int i = 0; i < 8; ++i) { b[i] = b0[i]; b[8 + i] = b1[i]; }
    acc = __builtin_amdgcn_wmma_f32_16x16x32_f16(
        false, a, false, b, (short)0, acc, false, false);
  }

  // C/D: VGPR r: lanes 0-15 -> M=row0+r; lanes 16-31 -> M=row0+8+r; N=col
  float* orow = hs + (size_t)(row0 + half * 8) * HFEAT + col;
  if (row0 + 16 <= n_rows) {                    // wave-uniform fast path
    #pragma unroll
    for (int r = 0; r < 8; ++r) orow[(size_t)r * HFEAT] = acc[r];
  } else {
    #pragma unroll
    for (int r = 0; r < 8; ++r) {
      int row = row0 + r + half * 8;
      if (row < n_rows) hs[(size_t)row * HFEAT + col] = acc[r];
    }
  }
}

// ---------------------------------------------------------------------------
// als[r] = dot(hs[r,:], a_s);  ald[r] = dot(X[r,:], wdad)   (one wave per row)
// ---------------------------------------------------------------------------
__global__ void gat_attn_terms(const float* __restrict__ hs,
                               const float* __restrict__ X,
                               const float* __restrict__ a_s,
                               const float* __restrict__ wdad,
                               float* __restrict__ als, float* __restrict__ ald,
                               int n, int din) {
  int gwave  = (blockIdx.x * blockDim.x + threadIdx.x) >> 5;
  int lane   = threadIdx.x & 31;
  int nwaves = (gridDim.x * blockDim.x) >> 5;
  for (int r = gwave; r < n; r += nwaves) {
    const float* hr = hs + (size_t)r * HFEAT;
    float s = hr[lane] * a_s[lane] + hr[32 + lane] * a_s[32 + lane];
    float d = 0.f;
    const float* xr = X + (size_t)r * din;
    for (int k = lane; k < din; k += 32) d += xr[k] * wdad[k];
    #pragma unroll
    for (int off = 16; off; off >>= 1) {
      s += __shfl_xor(s, off, 32);
      d += __shfl_xor(d, off, 32);
    }
    if (lane == 0) { als[r] = s; ald[r] = d; }
  }
}

// ---------------------------------------------------------------------------
// logit[e] = relu(als[src]+ald[dst]); segmax[dst] = max(...)   (int atomicMax
// on bit pattern is exact for non-negative f32; segmax zeroed, relu >= 0)
// ---------------------------------------------------------------------------
__global__ void gat_edge_logit(const int* __restrict__ src, const int* __restrict__ dst,
                               const float* __restrict__ als, const float* __restrict__ ald,
                               float* __restrict__ logit, float* __restrict__ segmax, int ne) {
  int e = blockIdx.x * blockDim.x + threadIdx.x;
  if (e >= ne) return;
  __builtin_prefetch(src + e + 16384, 0, 1);     // global_prefetch_b8 (speculative)
  __builtin_prefetch(dst + e + 16384, 0, 1);
  float l = als[src[e]] + ald[dst[e]];
  l = l > 0.f ? l : 0.f;
  logit[e] = l;
  atomicMax((int*)segmax + dst[e], __float_as_int(l));
}

// ---------------------------------------------------------------------------
// ex[e] = exp(logit - segmax[dst]);  denom[dst] += ex   (in-place on logit buf)
// ---------------------------------------------------------------------------
__global__ void gat_edge_exp(const int* __restrict__ dst,
                             const float* __restrict__ segmax,
                             float* __restrict__ exv, float* __restrict__ denom, int ne) {
  int e = blockIdx.x * blockDim.x + threadIdx.x;
  if (e >= ne) return;
  int d = dst[e];
  float ex = __expf(exv[e] - segmax[d]);
  exv[e] = ex;
  unsafeAtomicAdd(denom + d, ex);
}

// ---------------------------------------------------------------------------
// out[dst,:] += (ex/(denom+eps)) * hs[src,:]
// 16 threads per edge, float4 per thread: hs gather becomes global_load_b128,
// dst/ex/denom loads amortized 4x; same global_atomic_add_f32 count.
// ---------------------------------------------------------------------------
__global__ void gat_edge_aggr(const int* __restrict__ src, const int* __restrict__ dst,
                              const float* __restrict__ exv, const float* __restrict__ denom,
                              const float* __restrict__ hs, float* __restrict__ out, int ne) {
  long long idx = (long long)blockIdx.x * blockDim.x + threadIdx.x;
  int e  = (int)(idx >> 4);
  int f4 = (int)(idx & 15) << 2;
  if (e >= ne) return;
  int d = dst[e];
  float alpha = exv[e] / (denom[d] + EPSV);
  float4 hv = *(const float4*)(hs + (size_t)src[e] * HFEAT + f4);
  float* op = out + (size_t)d * HFEAT + f4;
  unsafeAtomicAdd(op + 0, alpha * hv.x);
  unsafeAtomicAdd(op + 1, alpha * hv.y);
  unsafeAtomicAdd(op + 2, alpha * hv.z);
  unsafeAtomicAdd(op + 3, alpha * hv.w);
}

__global__ void gat_bias(float* __restrict__ out, const float* __restrict__ b, int total) {
  int i = blockIdx.x * blockDim.x + threadIdx.x;
  if (i < total) out[i] += b[i & (HFEAT - 1)];
}

// ---------------------------------------------------------------------------
// Host-side: one GAT layer (all on `stream`, graph-capture safe)
// ---------------------------------------------------------------------------
static void run_gat_layer(const float* x, int din,
                          const float* Ws, const float* Wd,
                          const float* a_s, const float* a_d, const float* b,
                          float* out, float* hs, float* exv,
                          float* als, float* ald, float* segmax, float* denom, float* wdad,
                          const int* src, const int* dst, int n, int ne, hipStream_t stream) {
  gat_fuse_wdad<<<1, 128, 0, stream>>>(Wd, a_d, wdad, din);
  if (din == 128)
    gat_gemm_hs<128><<<(n + 31) / 32, 256, 0, stream>>>(x, Ws, hs, n);
  else
    gat_gemm_hs<64><<<(n + 31) / 32, 256, 0, stream>>>(x, Ws, hs, n);
  gat_attn_terms<<<512, 256, 0, stream>>>(hs, x, a_s, wdad, als, ald, n, din);
  hipMemsetAsync(segmax, 0, (size_t)n * sizeof(float), stream);
  hipMemsetAsync(denom, 0, (size_t)n * sizeof(float), stream);
  gat_edge_logit<<<(ne + 255) / 256, 256, 0, stream>>>(src, dst, als, ald, exv, segmax, ne);
  gat_edge_exp<<<(ne + 255) / 256, 256, 0, stream>>>(dst, segmax, exv, denom, ne);
  hipMemsetAsync(out, 0, (size_t)n * HFEAT * sizeof(float), stream);  // after attn_terms read x (may alias)
  gat_edge_aggr<<<(unsigned)(((long long)ne * 16 + 255) / 256), 256, 0, stream>>>(
      src, dst, exv, denom, hs, out, ne);
  gat_bias<<<(n * HFEAT + 255) / 256, 256, 0, stream>>>(out, b, n * HFEAT);
}

extern "C" void kernel_launch(void* const* d_in, const int* in_sizes, int n_in,
                              void* d_out, int out_size, void* d_ws, size_t ws_size,
                              hipStream_t stream) {
  const float* x  = (const float*)d_in[0];
  const int*   ei = (const int*)d_in[1];
  const int din0 = 128;
  const int n  = in_sizes[0] / din0;
  const int ne = in_sizes[1] / 2;
  const int* src = ei;
  const int* dst = ei + ne;

  const float* W[3][2]; const float* av[3][2]; const float* bb[3];
  for (int l = 0; l < 3; ++l) {
    W[l][0]  = (const float*)d_in[2 + l * 5 + 0];
    W[l][1]  = (const float*)d_in[2 + l * 5 + 1];
    av[l][0] = (const float*)d_in[2 + l * 5 + 2];
    av[l][1] = (const float*)d_in[2 + l * 5 + 3];
    bb[l]    = (const float*)d_in[2 + l * 5 + 4];
  }

  // Workspace partition (floats)
  float* p = (float*)d_ws;
  float* hs     = p;                    p += (size_t)n * HFEAT;
  float* bufA   = p;                    p += (size_t)n * HFEAT;
  float* exv    = p;                    p += (size_t)ne;
  float* als    = p;                    p += n;
  float* ald    = p;                    p += n;
  float* segmax = p;                    p += n;
  float* denom  = p;                    p += n;
  float* wdad   = p;                    p += 128;

  float* outf = (float*)d_out;

  // Layer 1: x[N,128] -> bufA[N,64]
  run_gat_layer(x, din0, W[0][0], W[0][1], av[0][0], av[0][1], bb[0],
                bufA, hs, exv, als, ald, segmax, denom, wdad, src, dst, n, ne, stream);
  // Layer 2: bufA -> bufA (x fully consumed before out is zeroed; stream-ordered)
  run_gat_layer(bufA, HFEAT, W[1][0], W[1][1], av[1][0], av[1][1], bb[1],
                bufA, hs, exv, als, ald, segmax, denom, wdad, src, dst, n, ne, stream);
  // Layer 3: bufA -> d_out
  run_gat_layer(bufA, HFEAT, W[2][0], W[2][1], av[2][0], av[2][1], bb[2],
                outf, hs, exv, als, ald, segmax, denom, wdad, src, dst, n, ne, stream);
}
